// TokenSwapMamba_16011638079732
// MI455X (gfx1250) — matrix operand order repaired and checked by reading, not compiled, not run on hardware
//
#include <hip/hip_runtime.h>
#include <hip/hip_bf16.h>

// ---------------------------------------------------------------------------
// TokenSwapMamba for MI455X (gfx1250, wave32, WMMA)
//   B=16, L=2048, DIM=256, D_INNER=512, D_STATE=16, D_CONV=4, DT_RANK=16
// ---------------------------------------------------------------------------

#define BATCH   16
#define SEQ     2048
#define DIM     256
#define DINNER  512
#define DSTATE  16
#define DTRANK  16
#define MROWS   (BATCH * SEQ)          // 32768

typedef __attribute__((ext_vector_type(16))) __bf16 v16bf;
typedef __attribute__((ext_vector_type(8)))  __bf16 v8bf;
typedef __attribute__((ext_vector_type(8)))  float  v8f;

// ---------------------------------------------------------------------------
// f32 -> bf16 conversion (weights / one-time)
// ---------------------------------------------------------------------------
__global__ void cvt_bf16_kernel(const float* __restrict__ src,
                                __hip_bfloat16* __restrict__ dst, int n) {
  int i = blockIdx.x * 256 + threadIdx.x;
  if (i < n) dst[i] = __float2bfloat16(src[i]);
}

// ---------------------------------------------------------------------------
// LayerNorm both inputs + channel swap, emit bf16 activations for GEMM1.
//   x1s[c] = c < nc ? x2n[c] : x1n[c];  x2s[c] = x2n[c] (after faithful 2-step swap)
// One block per (b,l) row, 256 threads.
// ---------------------------------------------------------------------------
__global__ __launch_bounds__(256)
void ln_swap_kernel(const float* __restrict__ x1, const float* __restrict__ x2,
                    const float* __restrict__ ratio,
                    const float* __restrict__ ln1w, const float* __restrict__ ln1b,
                    const float* __restrict__ ln2w, const float* __restrict__ ln2b,
                    __hip_bfloat16* __restrict__ xs1, __hip_bfloat16* __restrict__ xs2) {
  const int c   = threadIdx.x;
  const size_t row = blockIdx.x;
  __shared__ float red[256];
  __shared__ float stats[4];   // mu1, E[x1^2], mu2, E[x2^2]

  const float v1 = x1[row * DIM + c];
  const float v2 = x2[row * DIM + c];

  float vals[4] = {v1, v1 * v1, v2, v2 * v2};
  for (int q = 0; q < 4; ++q) {
    red[c] = vals[q];
    __syncthreads();
    for (int s = 128; s > 0; s >>= 1) {
      if (c < s) red[c] += red[c + s];
      __syncthreads();
    }
    if (c == 0) stats[q] = red[0] * (1.0f / DIM);
    __syncthreads();
  }
  const float mu1 = stats[0], var1 = stats[1] - mu1 * mu1;
  const float mu2 = stats[2], var2 = stats[3] - mu2 * mu2;

  const float x1n = (v1 - mu1) * __frsqrt_rn(var1 + 1e-5f) * ln1w[c] + ln1b[c];
  const float x2n = (v2 - mu2) * __frsqrt_rn(var2 + 1e-5f) * ln2w[c] + ln2b[c];

  const float r = ratio[0];
  int nc = (int)((float)DIM / (1.0f + __expf(-r)));
  if (nc < 0) nc = 0; if (nc > DIM) nc = DIM;

  xs1[row * DIM + c] = __float2bfloat16(c < nc ? x2n : x1n);
  xs2[row * DIM + c] = __float2bfloat16(x2n);
}

// ---------------------------------------------------------------------------
// bf16 WMMA GEMM:  C[M][N] = A[M][K] * W[N][K]^T  (+ optional f32 residual)
//
// Wave tile: (16*MT) x (16*NT); 4 waves/block along M.
// Software-pipelined: fragments for K-step k+32 are prefetched into a second
// register buffer before the current step's WMMAs issue, so the compiler can
// overlap global-load latency with matrix math (partial s_wait_loadcnt).
// Fragment layouts per CDNA5 ISA 7.12.2 (16-bit A 16x32, B 32x16, f32 C 16x16).
// ---------------------------------------------------------------------------
template <int NT, int MT, int K>
__global__ __launch_bounds__(128)
void gemm_bf16_kernel(const __bf16* __restrict__ A, const __bf16* __restrict__ W,
                      float* __restrict__ C, const float* __restrict__ resid,
                      int M, int N) {
  const int lane  = threadIdx.x & 31;
  const int wave  = threadIdx.x >> 5;
  const int hi    = lane >> 4;       // half-wave select
  const int l16   = lane & 15;
  const int mBase = (blockIdx.y * 4 + wave) * (16 * MT);
  const int nBase = blockIdx.x * (16 * NT);
  if (mBase >= M) return;

  v8f acc[MT][NT];
#pragma unroll
  for (int mt = 0; mt < MT; ++mt)
#pragma unroll
    for (int t = 0; t < NT; ++t)
#pragma unroll
      for (int i = 0; i < 8; ++i) acc[mt][t][i] = 0.0f;

  // Per-lane base pointers.
  const __bf16* aRow[MT];
#pragma unroll
  for (int mt = 0; mt < MT; ++mt)
    aRow[mt] = A + (size_t)(mBase + mt * 16 + l16) * K + hi * 8;
  const __bf16* wRow[NT];
#pragma unroll
  for (int t = 0; t < NT; ++t)
    wRow[t] = W + (size_t)(nBase + t * 16 + l16) * K + hi * 8;

  // Fragment loaders (two b128 loads each).
  auto loadA = [&](int mt, int k0) -> v16bf {
    const v8bf a0 = *(const v8bf*)(aRow[mt] + k0);
    const v8bf a1 = *(const v8bf*)(aRow[mt] + k0 + 16);
    v16bf f;
#pragma unroll
    for (int i = 0; i < 8; ++i) { f[i] = a0[i]; f[8 + i] = a1[i]; }
    return f;
  };
  auto loadB = [&](int t, int k0) -> v16bf {
    const v8bf b0 = *(const v8bf*)(wRow[t] + k0);
    const v8bf b1 = *(const v8bf*)(wRow[t] + k0 + 16);
    v16bf f;
#pragma unroll
    for (int i = 0; i < 8; ++i) { f[i] = b0[i]; f[8 + i] = b1[i]; }
    return f;
  };

  // Prologue: fetch K-step 0.
  v16bf aCur[MT], bCur[NT];
#pragma unroll
  for (int mt = 0; mt < MT; ++mt) aCur[mt] = loadA(mt, 0);
#pragma unroll
  for (int t = 0; t < NT; ++t) bCur[t] = loadB(t, 0);

#pragma unroll
  for (int k0 = 0; k0 < K; k0 += 32) {
    v16bf aNxt[MT], bNxt[NT];
    if (k0 + 32 < K) {
      // Prefetch next K-step BEFORE current WMMAs -> latency overlap.
#pragma unroll
      for (int mt = 0; mt < MT; ++mt) aNxt[mt] = loadA(mt, k0 + 32);
#pragma unroll
      for (int t = 0; t < NT; ++t) bNxt[t] = loadB(t, k0 + 32);
    }
#pragma unroll
    for (int mt = 0; mt < MT; ++mt)
#pragma unroll
      for (int t = 0; t < NT; ++t)
        acc[mt][t] = __builtin_amdgcn_wmma_f32_16x16x32_bf16(
            false, aCur[mt], false, bCur[t], (short)0, acc[mt][t], false, false);
    if (k0 + 32 < K) {
#pragma unroll
      for (int mt = 0; mt < MT; ++mt) aCur[mt] = aNxt[mt];
#pragma unroll
      for (int t = 0; t < NT; ++t) bCur[t] = bNxt[t];
    }
  }

  // C layout: VGPR r -> m = mBase + mt*16 + r + 8*hi, n = nBase + t*16 + (lane&15)
#pragma unroll
  for (int mt = 0; mt < MT; ++mt)
#pragma unroll
    for (int t = 0; t < NT; ++t) {
      const int n = nBase + t * 16 + l16;
#pragma unroll
      for (int r = 0; r < 8; ++r) {
        const int m = mBase + mt * 16 + r + 8 * hi;
        const size_t idx = (size_t)m * N + n;
        float v = acc[mt][t][r];
        if (resid) v += resid[idx];
        C[idx] = v;
      }
    }
}

// ---------------------------------------------------------------------------
// Causal depthwise conv1d (k=4, left pad 3) + SiLU over the u half of xz.
// xz: [M][2*DINNER] f32 ; outputs u_f32 [M][DINNER] and u_bf16 for GEMM2.
// ---------------------------------------------------------------------------
__global__ __launch_bounds__(256)
void conv_silu_kernel(const float* __restrict__ xz, const float* __restrict__ cw,
                      const float* __restrict__ cb, float* __restrict__ uf,
                      __hip_bfloat16* __restrict__ ubf) {
  const int i = blockIdx.x * 256 + threadIdx.x;           // over M*DINNER
  const int d = i & (DINNER - 1);
  const size_t m = (size_t)(i >> 9);
  const int l = (int)(m & (SEQ - 1));
  float acc = cb[d];
#pragma unroll
  for (int j = 0; j < 4; ++j) {
    const int ll = l - 3 + j;
    if (ll >= 0) acc += xz[(m - (size_t)(3 - j)) * (2 * DINNER) + d] * cw[d * 4 + j];
  }
  const float s = acc / (1.0f + __expf(-acc));            // SiLU
  uf[i] = s;
  ubf[i] = __float2bfloat16(s);
}

// ---------------------------------------------------------------------------
// delta = softplus(dt @ dt_w^T + dt_b); dt = dbl[:, 0:16]
// ---------------------------------------------------------------------------
__global__ __launch_bounds__(256)
void delta_kernel(const float* __restrict__ dbl, const float* __restrict__ dtw,
                  const float* __restrict__ dtb, float* __restrict__ delta) {
  const int i = blockIdx.x * 256 + threadIdx.x;           // over M*DINNER
  const int d = i & (DINNER - 1);
  const size_t m = (size_t)(i >> 9);
  float acc = dtb[d];
#pragma unroll
  for (int r = 0; r < DTRANK; ++r)
    acc += dbl[m * 48 + r] * dtw[d * DTRANK + r];
  const float sp = (acc > 20.0f) ? acc : log1pf(__expf(acc));
  delta[i] = sp;
}

// ---------------------------------------------------------------------------
// Selective scan, serial in L, parallel over (b, d). h[16] in VGPRs,
// per-step B/C (32 floats) broadcast through LDS. Fused epilogue:
//   y = (scan_y + u*Dp) * silu(z) -> bf16 for the out-proj GEMM.
// ---------------------------------------------------------------------------
__global__ __launch_bounds__(512)
void scan_kernel(const float* __restrict__ delta, const float* __restrict__ u,
                 const float* __restrict__ dbl, const float* __restrict__ xz,
                 const float* __restrict__ A_log, const float* __restrict__ Dp,
                 __hip_bfloat16* __restrict__ ybf) {
  const int b = blockIdx.x;
  const int d = threadIdx.x;
  float A[DSTATE], h[DSTATE];
#pragma unroll
  for (int s = 0; s < DSTATE; ++s) {
    A[s] = -__expf(A_log[d * DSTATE + s]);
    h[s] = 0.0f;
  }
  const float dpv = Dp[d];
  __shared__ float BC[32];

  for (int l = 0; l < SEQ; ++l) {
    const size_t m = (size_t)b * SEQ + l;
    __syncthreads();
    if (d < 32) BC[d] = dbl[m * 48 + DTRANK + d];   // Bm (16) then Cm (16)
    __syncthreads();
    const float dl = delta[m * DINNER + d];
    const float uu = u[m * DINNER + d];
    const float du = dl * uu;
    float y = 0.0f;
#pragma unroll
    for (int s = 0; s < DSTATE; ++s) {
      h[s] = h[s] * __expf(dl * A[s]) + du * BC[s];
      y += h[s] * BC[DSTATE + s];
    }
    const float z  = xz[m * (2 * DINNER) + DINNER + d];
    const float sz = z / (1.0f + __expf(-z));
    ybf[m * DINNER + d] = __float2bfloat16((y + uu * dpv) * sz);
  }
}

// ---------------------------------------------------------------------------
// Host-side launch
// ---------------------------------------------------------------------------
extern "C" void kernel_launch(void* const* d_in, const int* in_sizes, int n_in,
                              void* d_out, int out_size, void* d_ws, size_t ws_size,
                              hipStream_t stream) {
  (void)in_sizes; (void)n_in; (void)out_size; (void)ws_size;
  const float* x_in[2]  = {(const float*)d_in[0], (const float*)d_in[1]};
  const float* ratio    = (const float*)d_in[2];
  const float* ln_w[2]  = {(const float*)d_in[3], (const float*)d_in[5]};
  const float* ln_b[2]  = {(const float*)d_in[4], (const float*)d_in[6]};

  // per-stream parameter bases: e1 at 7, e2 at 16
  const float *in_w[2], *conv_w[2], *conv_b[2], *xproj_w[2], *dt_w[2], *dt_b[2];
  const float *A_log[2], *Dp[2], *out_w[2];
  for (int s = 0; s < 2; ++s) {
    const int base = 7 + 9 * s;
    in_w[s]    = (const float*)d_in[base + 0];
    conv_w[s]  = (const float*)d_in[base + 1];
    conv_b[s]  = (const float*)d_in[base + 2];
    xproj_w[s] = (const float*)d_in[base + 3];
    dt_w[s]    = (const float*)d_in[base + 4];
    dt_b[s]    = (const float*)d_in[base + 5];
    A_log[s]   = (const float*)d_in[base + 6];
    Dp[s]      = (const float*)d_in[base + 7];
    out_w[s]   = (const float*)d_in[base + 8];
  }

  // ---- workspace carve-out ------------------------------------------------
  char* cur = (char*)d_ws;
  auto carve = [&](size_t bytes) -> void* {
    void* r = (void*)cur;
    cur += (bytes + 255) & ~(size_t)255;
    return r;
  };
  __hip_bfloat16 *xs_bf[2], *u_bf[2], *y_bf[2], *inw_bf[2], *xpw_bf[2], *outw_bf[2];
  float *xz[2], *u_f[2], *dbl[2], *delta[2];
  for (int s = 0; s < 2; ++s) {
    xs_bf[s]   = (__hip_bfloat16*)carve((size_t)MROWS * DIM * 2);
    u_bf[s]    = (__hip_bfloat16*)carve((size_t)MROWS * DINNER * 2);
    y_bf[s]    = (__hip_bfloat16*)carve((size_t)MROWS * DINNER * 2);
    inw_bf[s]  = (__hip_bfloat16*)carve((size_t)(2 * DINNER) * DIM * 2);
    xpw_bf[s]  = (__hip_bfloat16*)carve((size_t)48 * DINNER * 2);
    outw_bf[s] = (__hip_bfloat16*)carve((size_t)DIM * DINNER * 2);
    xz[s]      = (float*)carve((size_t)MROWS * 2 * DINNER * 4);
    u_f[s]     = (float*)carve((size_t)MROWS * DINNER * 4);
    dbl[s]     = (float*)carve((size_t)MROWS * 48 * 4);
    delta[s]   = (float*)carve((size_t)MROWS * DINNER * 4);
  }

  // ---- 1) weight conversion to bf16 --------------------------------------
  auto cvt = [&](const float* src, __hip_bfloat16* dst, int n) {
    cvt_bf16_kernel<<<(n + 255) / 256, 256, 0, stream>>>(src, dst, n);
  };
  for (int s = 0; s < 2; ++s) {
    cvt(in_w[s],    inw_bf[s],  2 * DINNER * DIM);
    cvt(xproj_w[s], xpw_bf[s],  48 * DINNER);
    cvt(out_w[s],   outw_bf[s], DIM * DINNER);
  }

  // ---- 2) layernorm + channel swap ---------------------------------------
  ln_swap_kernel<<<MROWS, 256, 0, stream>>>(
      x_in[0], x_in[1], ratio, ln_w[0], ln_b[0], ln_w[1], ln_b[1],
      xs_bf[0], xs_bf[1]);

  // ---- 3..8) two Mamba pipelines ------------------------------------------
  const int elemGrid = (MROWS * DINNER) / 256;
  const int mBlocks2 = MROWS / (16 * 2 * 4);   // MT=2, 4 waves/block => 256
  for (int s = 0; s < 2; ++s) {
    // in-proj: xz = xs @ in_w^T   (M=32768, N=1024, K=256)
    gemm_bf16_kernel<4, 2, DIM><<<dim3((2 * DINNER) / 64, mBlocks2), 128, 0, stream>>>(
        (const __bf16*)xs_bf[s], (const __bf16*)inw_bf[s], xz[s], nullptr,
        MROWS, 2 * DINNER);

    // depthwise conv + SiLU
    conv_silu_kernel<<<elemGrid, 256, 0, stream>>>(
        xz[s], conv_w[s], conv_b[s], u_f[s], u_bf[s]);

    // x-proj: dbl = u @ xproj_w^T (M=32768, N=48, K=512)
    gemm_bf16_kernel<3, 2, DINNER><<<dim3(1, mBlocks2), 128, 0, stream>>>(
        (const __bf16*)u_bf[s], (const __bf16*)xpw_bf[s], dbl[s], nullptr,
        MROWS, 48);

    // delta = softplus(dt @ dt_w^T + dt_b)
    delta_kernel<<<elemGrid, 256, 0, stream>>>(dbl[s], dt_w[s], dt_b[s], delta[s]);

    // selective scan (serial in L) + gating epilogue
    scan_kernel<<<BATCH, DINNER, 0, stream>>>(
        delta[s], u_f[s], dbl[s], xz[s], A_log[s], Dp[s], y_bf[s]);

    // out-proj: o = y @ out_w^T + x_short  (M=32768, N=256, K=512)
    float* out_s = (float*)d_out + (size_t)s * MROWS * DIM;
    gemm_bf16_kernel<4, 2, DINNER><<<dim3(DIM / 64, mBlocks2), 128, 0, stream>>>(
        (const __bf16*)y_bf[s], (const __bf16*)outw_bf[s], out_s, x_in[s],
        MROWS, DIM);
  }
}
